// CausalAttention_884763263241
// MI455X (gfx1250) — matrix operand order repaired and checked by reading, not compiled, not run on hardware
//
#include <hip/hip_runtime.h>
#include <type_traits>

#define TT 2048
#define DD 1024
#define BB 4

typedef __attribute__((ext_vector_type(8)))  __bf16        v8bf;
typedef __attribute__((ext_vector_type(16))) __bf16        v16bf;
typedef __attribute__((ext_vector_type(8)))  float         v8f;
typedef __attribute__((ext_vector_type(8)))  unsigned short u16x8;

__device__ __forceinline__ unsigned short f2bf(float f) {
    union { float f; unsigned u; } c; c.f = f;
    unsigned u = c.u + 0x7FFFu + ((c.u >> 16) & 1u);   // round-to-nearest-even
    return (unsigned short)(u >> 16);
}

// A-fragment (16x32, 16-bit): lane holds row M=lane&15.
// element e in [0,8) -> K = e + 8*hi ; e in [8,16) -> K = e + 8 + 8*hi
__device__ __forceinline__ v16bf load_fragA(const unsigned short* p32, int hi) {
    const v8bf lo = *(const v8bf*)(p32 + 8 * hi);
    const v8bf up = *(const v8bf*)(p32 + 16 + 8 * hi);
    return __builtin_shufflevector(lo, up, 0,1,2,3,4,5,6,7,8,9,10,11,12,13,14,15);
}

// B-fragment (32x16, 16-bit): lane holds column N=lane&15.
// element e in [0,8) -> K = e + 8*hi ; e in [8,16) -> K = e + 8 + 8*hi
__device__ __forceinline__ v16bf load_fragB(const unsigned short* p32, int hi) {
    const v8bf lo = *(const v8bf*)(p32 + 8 * hi);
    const v8bf up = *(const v8bf*)(p32 + 16 + 8 * hi);
    return __builtin_shufflevector(lo, up, 0,1,2,3,4,5,6,7,8,9,10,11,12,13,14,15);
}

__device__ __forceinline__ float rmax16(float v) {
    v = fmaxf(v, __shfl_xor(v, 1, 32));
    v = fmaxf(v, __shfl_xor(v, 2, 32));
    v = fmaxf(v, __shfl_xor(v, 4, 32));
    v = fmaxf(v, __shfl_xor(v, 8, 32));
    return v;
}
__device__ __forceinline__ float rsum16(float v) {
    v += __shfl_xor(v, 1, 32);
    v += __shfl_xor(v, 2, 32);
    v += __shfl_xor(v, 4, 32);
    v += __shfl_xor(v, 8, 32);
    return v;
}

// ---------------------------------------------------------------- convert
__global__ __launch_bounds__(256) void cvt_f32_bf16(const float* __restrict__ in,
                                                    unsigned short* __restrict__ out,
                                                    int n) {
    int i = blockIdx.x * 256 + threadIdx.x;
    if (i < n) out[i] = f2bf(in[i]);
}

// ---------------------------------------------------------------- GEMM
// C[M,N] = A[M,K] * B[K,N], bf16 inputs, fp32 accumulate.
// 128x128 tile per workgroup, 8 waves (4x2), each wave 32x64 = 2x4 WMMA tiles.
// Double-buffered LDS (ping-pong): tile t+1 is prefetched from global into
// registers before the WMMAs of tile t, written to the alternate buffer after,
// single barrier per k-step.
// causal!=0 clips the k loop at m0+128 (used for P*V, P is lower-triangular).
template <typename OutT>
__global__ __launch_bounds__(256) void gemm_wmma_bf16(
    const unsigned short* __restrict__ A, long strideA,
    const unsigned short* __restrict__ Bm, long strideB,
    OutT* __restrict__ C, long strideC,
    int K, int lda, int ldb, int ldc,
    float scale, int causal) {
    __shared__ unsigned short As[2][128 * 40];   // row-major, stride 40 halves
    __shared__ unsigned short Bt[2][128 * 40];   // B transposed: Bt[n][k]

    const int bz = blockIdx.z;
    A  += (size_t)bz * strideA;
    Bm += (size_t)bz * strideB;
    C  += (size_t)bz * strideC;

    const int m0 = blockIdx.y * 128, n0 = blockIdx.x * 128;
    const int tid = threadIdx.x, lane = tid & 31, wave = tid >> 5;
    const int hi = (lane >> 4) & 1, l16 = lane & 15;
    const int wr = wave >> 1, wc = wave & 1;

    // per-thread staging coordinates (2 slots each for A and B tiles)
    const int arow0 = tid >> 2,           aseg0 = tid & 3;
    const int arow1 = (tid + 256) >> 2,   aseg1 = tid & 3;
    const int bkk0  = tid >> 4,           bseg0 = tid & 15;
    const int bkk1  = (tid + 256) >> 4,   bseg1 = tid & 15;

    v8f acc[2][4];
#pragma unroll
    for (int i = 0; i < 2; ++i)
#pragma unroll
        for (int j = 0; j < 4; ++j)
#pragma unroll
            for (int r = 0; r < 8; ++r) acc[i][j][r] = 0.f;

    const int kBound = causal ? ((K < m0 + 128) ? K : (m0 + 128)) : K;
    const int nk = kBound >> 5;                 // #32-deep k-tiles (>=1)

    // ---- prologue: stage tile 0 into buffer 0
    {
        u16x8 a0 = *(const u16x8*)&A[(size_t)(m0 + arow0) * lda + aseg0 * 8];
        u16x8 a1 = *(const u16x8*)&A[(size_t)(m0 + arow1) * lda + aseg1 * 8];
        u16x8 b0 = *(const u16x8*)&Bm[(size_t)bkk0 * ldb + n0 + bseg0 * 8];
        u16x8 b1 = *(const u16x8*)&Bm[(size_t)bkk1 * ldb + n0 + bseg1 * 8];
        *(u16x8*)&As[0][arow0 * 40 + aseg0 * 8] = a0;
        *(u16x8*)&As[0][arow1 * 40 + aseg1 * 8] = a1;
#pragma unroll
        for (int t = 0; t < 8; ++t) Bt[0][(bseg0 * 8 + t) * 40 + bkk0] = b0[t];
#pragma unroll
        for (int t = 0; t < 8; ++t) Bt[0][(bseg1 * 8 + t) * 40 + bkk1] = b1[t];
    }
    __syncthreads();

    for (int t = 0; t < nk; ++t) {
        const int cur = t & 1, nxt = cur ^ 1;
        const bool hasNext = (t + 1 < nk);

        // prefetch tile t+1 from global into registers (overlaps the WMMAs)
        u16x8 pa0, pa1, pb0, pb1;
        if (hasNext) {
            const int k0n = (t + 1) << 5;
            pa0 = *(const u16x8*)&A[(size_t)(m0 + arow0) * lda + k0n + aseg0 * 8];
            pa1 = *(const u16x8*)&A[(size_t)(m0 + arow1) * lda + k0n + aseg1 * 8];
            pb0 = *(const u16x8*)&Bm[(size_t)(k0n + bkk0) * ldb + n0 + bseg0 * 8];
            pb1 = *(const u16x8*)&Bm[(size_t)(k0n + bkk1) * ldb + n0 + bseg1 * 8];
        }

        // compute on buffer `cur`
        v16bf a[2], bb[4];
#pragma unroll
        for (int i = 0; i < 2; ++i)
            a[i] = load_fragA(&As[cur][(wr * 32 + i * 16 + l16) * 40], hi);
#pragma unroll
        for (int j = 0; j < 4; ++j)
            bb[j] = load_fragB(&Bt[cur][(wc * 64 + j * 16 + l16) * 40], hi);
#pragma unroll
        for (int i = 0; i < 2; ++i)
#pragma unroll
            for (int j = 0; j < 4; ++j)
                acc[i][j] = __builtin_amdgcn_wmma_f32_16x16x32_bf16(
                    false, a[i], false, bb[j], (short)0, acc[i][j], false, false);

        // commit prefetched tile into the alternate buffer
        if (hasNext) {
            *(u16x8*)&As[nxt][arow0 * 40 + aseg0 * 8] = pa0;
            *(u16x8*)&As[nxt][arow1 * 40 + aseg1 * 8] = pa1;
#pragma unroll
            for (int q = 0; q < 8; ++q) Bt[nxt][(bseg0 * 8 + q) * 40 + bkk0] = pb0[q];
#pragma unroll
            for (int q = 0; q < 8; ++q) Bt[nxt][(bseg1 * 8 + q) * 40 + bkk1] = pb1[q];
        }
        __syncthreads();
    }

#pragma unroll
    for (int i = 0; i < 2; ++i)
#pragma unroll
        for (int j = 0; j < 4; ++j)
#pragma unroll
            for (int r = 0; r < 8; ++r) {
                int row = m0 + wr * 32 + i * 16 + r + 8 * hi;
                int col = n0 + wc * 64 + j * 16 + l16;
                float v = acc[i][j][r] * scale;
                if constexpr (std::is_same<OutT, float>::value)
                    C[(size_t)row * ldc + col] = v;
                else
                    C[(size_t)row * ldc + col] = f2bf(v);
            }
}

// ---------------------------------------------------------------- softmax(QK^T)
// 4 waves/WG; wave owns 16 query rows (Q held as 32 A-frags in VGPRs).
// Pass 1: online (m,l). Pass 2: write P = exp(S-m)/l (bf16), including the
// zero band up to the next 128-row boundary (what the PV GEMM will read).
__global__ __launch_bounds__(128) void attn_softmax_p(
    const unsigned short* __restrict__ Q,
    const unsigned short* __restrict__ Kmat,
    unsigned short* __restrict__ P) {
    __shared__ unsigned short Ks[16 * 1032];   // 16 key rows, stride 1032 halves

    const int wg = blockIdx.x;                 // 0..127
    const int b = wg >> 5;
    const int q0 = (wg & 31) * 64;
    const int tid = threadIdx.x, lane = tid & 31, wave = tid >> 5;
    const int hi = (lane >> 4) & 1, l16 = lane & 15;
    const int r0 = q0 + wave * 16;

    const unsigned short* Qb = Q + (size_t)b * TT * DD;
    const unsigned short* Kb = Kmat + (size_t)b * TT * DD;
    unsigned short* Pb = P + (size_t)b * TT * TT;

    // Q fragments for this wave's 16 rows: full K=1024 = 32 frags
    v16bf q[32];
#pragma unroll
    for (int f = 0; f < 32; ++f)
        q[f] = load_fragA(&Qb[(size_t)(r0 + l16) * DD + f * 32], hi);

    float mrun[8], lrun[8];
#pragma unroll
    for (int r = 0; r < 8; ++r) { mrun[r] = -INFINITY; lrun[r] = 0.f; }

    const int bound1 = q0 + 64;                        // covers all unmasked keys
    const int bound2 = ((q0 >> 7) << 7) + 128;         // P band read by PV GEMM

    // ---- pass 1: stats
    for (int j0 = 0; j0 < bound1; j0 += 16) {
#pragma unroll
        for (int it = 0; it < 16; ++it) {
            int slot = tid + it * 128;                 // 0..2047
            int row = slot >> 7, seg = slot & 127;
            *(u16x8*)&Ks[row * 1032 + seg * 8] =
                *(const u16x8*)&Kb[(size_t)(j0 + row) * DD + seg * 8];
        }
        __syncthreads();

        v8f s;
#pragma unroll
        for (int r = 0; r < 8; ++r) s[r] = 0.f;
#pragma unroll
        for (int f = 0; f < 32; ++f) {
            v16bf kf = load_fragB(&Ks[l16 * 1032 + f * 32], hi);
            s = __builtin_amdgcn_wmma_f32_16x16x32_bf16(
                false, q[f], false, kf, (short)0, s, false, false);
        }

        int col = j0 + l16;
#pragma unroll
        for (int r = 0; r < 8; ++r) {
            int row = r0 + r + 8 * hi;
            float sv = (col <= row) ? s[r] : -INFINITY;
            float tm = rmax16(sv);
            float nm = fmaxf(mrun[r], tm);
            float corr = __expf(mrun[r] - nm);
            float p = __expf(sv - nm);
            float ts = rsum16(p);
            lrun[r] = lrun[r] * corr + ts;
            mrun[r] = nm;
        }
        __syncthreads();
    }

    float linv[8];
#pragma unroll
    for (int r = 0; r < 8; ++r) linv[r] = 1.f / lrun[r];

    // ---- pass 2: write P
    for (int j0 = 0; j0 < bound2; j0 += 16) {
#pragma unroll
        for (int it = 0; it < 16; ++it) {
            int slot = tid + it * 128;
            int row = slot >> 7, seg = slot & 127;
            *(u16x8*)&Ks[row * 1032 + seg * 8] =
                *(const u16x8*)&Kb[(size_t)(j0 + row) * DD + seg * 8];
        }
        __syncthreads();

        v8f s;
#pragma unroll
        for (int r = 0; r < 8; ++r) s[r] = 0.f;
#pragma unroll
        for (int f = 0; f < 32; ++f) {
            v16bf kf = load_fragB(&Ks[l16 * 1032 + f * 32], hi);
            s = __builtin_amdgcn_wmma_f32_16x16x32_bf16(
                false, q[f], false, kf, (short)0, s, false, false);
        }

        int col = j0 + l16;
#pragma unroll
        for (int r = 0; r < 8; ++r) {
            int row = r0 + r + 8 * hi;
            float sv = (col <= row) ? s[r] : -INFINITY;
            float p = __expf(sv - mrun[r]) * linv[r];   // 0 where masked
            Pb[(size_t)row * TT + col] = f2bf(p);
        }
        __syncthreads();
    }
}

// ---------------------------------------------------------------- launch
extern "C" void kernel_launch(void* const* d_in, const int* in_sizes, int n_in,
                              void* d_out, int out_size, void* d_ws, size_t ws_size,
                              hipStream_t stream) {
    const float* x  = (const float*)d_in[0];
    const float* Wq = (const float*)d_in[1];
    const float* Wk = (const float*)d_in[2];
    const float* Wv = (const float*)d_in[3];
    float* out = (float*)d_out;

    char* ws = (char*)d_ws;
    const size_t XB = (size_t)BB * TT * DD;       // 8388608 elems
    const size_t WB = (size_t)DD * DD;            // 1048576 elems
    // byte offsets (all 16B-aligned); total ~107 MB of workspace
    unsigned short* xb  = (unsigned short*)(ws);
    unsigned short* wqb = (unsigned short*)(ws + 16777216);
    unsigned short* wkb = (unsigned short*)(ws + 18874368);
    unsigned short* wvb = (unsigned short*)(ws + 20971520);
    unsigned short* qb  = (unsigned short*)(ws + 23068672);
    unsigned short* kb  = (unsigned short*)(ws + 39845888);
    unsigned short* vb  = (unsigned short*)(ws + 56623104);
    unsigned short* pm  = (unsigned short*)(ws + 73400320);  // + 33554432 -> end

    // 1) fp32 -> bf16
    cvt_f32_bf16<<<(int)(XB / 256), 256, 0, stream>>>(x, xb, (int)XB);
    cvt_f32_bf16<<<(int)(WB / 256), 256, 0, stream>>>(Wq, wqb, (int)WB);
    cvt_f32_bf16<<<(int)(WB / 256), 256, 0, stream>>>(Wk, wkb, (int)WB);
    cvt_f32_bf16<<<(int)(WB / 256), 256, 0, stream>>>(Wv, wvb, (int)WB);

    // 2) Q/K/V projections: [8192x1024] @ [1024x1024]; 1/sqrt(1024) folded into Q
    dim3 gp(DD / 128, (BB * TT) / 128, 1);
    gemm_wmma_bf16<unsigned short><<<gp, 256, 0, stream>>>(
        xb, 0, wqb, 0, qb, 0, DD, DD, DD, DD, 1.0f / 32.0f, 0);
    gemm_wmma_bf16<unsigned short><<<gp, 256, 0, stream>>>(
        xb, 0, wkb, 0, kb, 0, DD, DD, DD, DD, 1.0f, 0);
    gemm_wmma_bf16<unsigned short><<<gp, 256, 0, stream>>>(
        xb, 0, wvb, 0, vb, 0, DD, DD, DD, DD, 1.0f, 0);

    // 3) causal softmax weights P (bf16), per 64-query-row workgroup
    attn_softmax_p<<<BB * (TT / 64), 128, 0, stream>>>(qb, kb, pm);

    // 4) O = P @ V per batch, fp32 out; causal clip of k loop
    dim3 go(DD / 128, TT / 128, BB);
    gemm_wmma_bf16<float><<<go, 256, 0, stream>>>(
        pm, (long)TT * TT, vb, (long)TT * DD, out, (long)TT * DD,
        TT, TT, DD, DD, 1.0f, 1);
}
